// CausalSelfAttention_80487687127241
// MI455X (gfx1250) — matrix operand order repaired and checked by reading, not compiled, not run on hardware
//
#include <hip/hip_runtime.h>
#include <math.h>

// ---------------- problem constants ----------------
constexpr int Bc = 2;
constexpr int Tc = 2048;
constexpr int Dc = 1024;
constexpr int Hc = 16;
constexpr int HDc = 64;          // head dim
constexpr int D3 = 3 * Dc;       // 3072
constexpr float SCALE = 0.125f;  // 1/sqrt(64)

typedef __attribute__((ext_vector_type(16))) __bf16 v16bf;
typedef __attribute__((ext_vector_type(8)))  float  v8f;
typedef unsigned short u16;

union V16U {
    v16bf v;
    uint4 u[2];
};

__device__ __forceinline__ u16 f32_to_bf16(float f) {
    union { float f; unsigned int u; } x;
    x.f = f;
    unsigned int u = x.u;
    u += 0x7FFFu + ((u >> 16) & 1u);   // round-to-nearest-even
    return (u16)(u >> 16);
}

__device__ __forceinline__ v8f v8f_zero() {
    v8f z = {0.f, 0.f, 0.f, 0.f, 0.f, 0.f, 0.f, 0.f};
    return z;
}

__device__ __forceinline__ v8f bf16_wmma(const V16U& a, const V16U& b, v8f c) {
    return __builtin_amdgcn_wmma_f32_16x16x32_bf16(false, a.v, false, b.v,
                                                   (short)0, c, false, false);
}

// ---------------- fp32 -> bf16 conversion ----------------
__global__ void cvt_f32_bf16_kernel(const float* __restrict__ in,
                                    u16* __restrict__ out, unsigned n) {
    unsigned i = blockIdx.x * blockDim.x + threadIdx.x;
    unsigned stride = gridDim.x * blockDim.x;
    for (; i < n; i += stride) out[i] = f32_to_bf16(in[i]);
}

// ---------------- V transpose: qkv[b,t,2D + h*64+d] -> vt[(b*H+h)*HD+d, t] ----------------
__global__ void vtrans_kernel(const u16* __restrict__ qkv, u16* __restrict__ vt) {
    unsigned total = (unsigned)Bc * Tc * Dc;
    unsigned i = blockIdx.x * blockDim.x + threadIdx.x;
    unsigned stride = gridDim.x * blockDim.x;
    for (; i < total; i += stride) {
        unsigned d = i % Dc;
        unsigned t = (i / Dc) % Tc;
        unsigned b = i / (Dc * Tc);
        unsigned h = d / HDc, dd = d % HDc;
        vt[((size_t)((b * Hc + h) * HDc + dd)) * Tc + t] =
            qkv[(size_t)(b * Tc + t) * D3 + 2 * Dc + d];
    }
}

// ---------------- GEMM: C[M,N] = A[M,K] (bf16) x W[N,K]^T (bf16), fp32 acc ----------------
// One wave computes a 32x64 C tile (2 M-tiles x 4 N-tiles = 8 WMMAs per K-step of 32).
// K-loop unrolled x2 with ping-pong register buffers so loads overlap WMMAs.
// __launch_bounds__(256,1): relax the VGPR budget so nothing spills to scratch.
template <bool OUT_BF16>
__global__ void __launch_bounds__(256, 1)
gemm_wmma_bf16(const u16* __restrict__ A, const u16* __restrict__ W,
               void* __restrict__ Cout, int M, int N, int K) {
    const int lane = threadIdx.x & 31;
    const int wave = threadIdx.x >> 5;
    const int tilesN = N >> 6;                       // 64-wide N tiles
    const int gid = blockIdx.x * (blockDim.x >> 5) + wave;
    if (gid >= (M >> 5) * tilesN) return;            // wave-uniform branch
    const int tm = gid / tilesN, tn = gid % tilesN;
    const int nm = lane & 15;
    const int half = lane >> 4;

    const u16* a0 = A + (size_t)(tm * 32 + nm) * K;
    const u16* a1 = a0 + (size_t)16 * K;
    const u16* w0 = W + (size_t)(tn * 64 + nm) * K;

    v8f c[2][4];
#pragma unroll
    for (int i = 0; i < 2; ++i)
#pragma unroll
        for (int j = 0; j < 4; ++j) c[i][j] = v8f_zero();

    V16U a[2], b[4], an[2], bn[4];

#define LOAD_A(dst, kk)                                                \
    do {                                                               \
        (dst)[0].u[0] = *(const uint4*)(a0 + (kk) + 8 * half);         \
        (dst)[0].u[1] = *(const uint4*)(a0 + (kk) + 16 + 8 * half);    \
        (dst)[1].u[0] = *(const uint4*)(a1 + (kk) + 8 * half);         \
        (dst)[1].u[1] = *(const uint4*)(a1 + (kk) + 16 + 8 * half);    \
    } while (0)
#define LOAD_B(dst, kk)                                                \
    do {                                                               \
        _Pragma("unroll")                                              \
        for (int j = 0; j < 4; ++j) {                                  \
            const u16* wr = w0 + (size_t)(j * 16) * K + (kk);          \
            (dst)[j].u[0] = *(const uint4*)(wr + 8 * half);            \
            (dst)[j].u[1] = *(const uint4*)(wr + 16 + 8 * half);       \
        }                                                              \
    } while (0)
#define MMA8(aa, bb)                                                   \
    do {                                                               \
        _Pragma("unroll")                                              \
        for (int j = 0; j < 4; ++j) {                                  \
            c[0][j] = bf16_wmma((aa)[0], (bb)[j], c[0][j]);            \
            c[1][j] = bf16_wmma((aa)[1], (bb)[j], c[1][j]);            \
        }                                                              \
    } while (0)

    LOAD_A(a, 0);
    LOAD_B(b, 0);
    for (int kk = 0; kk < K; kk += 64) {            // K % 64 == 0
        LOAD_A(an, kk + 32);                         // prefetch next half-step
        LOAD_B(bn, kk + 32);
        MMA8(a, b);                                  // consume current regs
        if (kk + 64 < K) {
            LOAD_A(a, kk + 64);
            LOAD_B(b, kk + 64);
        }
        MMA8(an, bn);
    }
#undef LOAD_A
#undef LOAD_B
#undef MMA8

    // C/D layout: VGPR r -> row r + 8*(lane/16), col lane%16
#pragma unroll
    for (int i = 0; i < 2; ++i)
#pragma unroll
        for (int j = 0; j < 4; ++j)
#pragma unroll
            for (int r = 0; r < 8; ++r) {
                size_t idx = (size_t)(tm * 32 + i * 16 + r + 8 * half) * N +
                             (tn * 64 + j * 16 + nm);
                if (OUT_BF16) ((u16*)Cout)[idx] = f32_to_bf16(c[i][j][r]);
                else          ((float*)Cout)[idx] = c[i][j][r];
            }
}

// ---------------- Flash attention: one wave per (b, h, 32-row q tile) ----------------
// 32 q-rows x 32 kv-cols per step: 8 score WMMAs + 8 PV WMMAs, K/V operands reused
// across both 16-row M-tiles. __launch_bounds__(256,1) keeps ~200 VGPRs resident.
#define AT_WAVES 8
__global__ void __launch_bounds__(256, 1)
flash_attn_kernel(const u16* __restrict__ qkv,
                  const u16* __restrict__ vt,
                  u16* __restrict__ zout) {
    __shared__ __align__(16) u16 Pbuf[AT_WAVES][2][16][32];

    const int lane = threadIdx.x & 31;
    const int wave = threadIdx.x >> 5;
    const int QT = Tc / 32;  // 64 q-tiles
    const int gid = blockIdx.x * AT_WAVES + wave;
    if (gid >= Bc * Hc * QT) return;               // wave-uniform
    const int b = gid / (Hc * QT);
    const int h = (gid / QT) % Hc;
    const int q0 = (gid % QT) * 32;
    const int nm = lane & 15;
    const int g = lane >> 4;

    // Q A-operands: 2 M-tiles x 2 d-chunks, held for the whole kv loop
    V16U aQ[2][2];
#pragma unroll
    for (int mt = 0; mt < 2; ++mt) {
        const u16* qrow = qkv + (size_t)(b * Tc + q0 + mt * 16 + nm) * D3 + h * HDc;
#pragma unroll
        for (int dc = 0; dc < 2; ++dc) {
            aQ[mt][dc].u[0] = *(const uint4*)(qrow + 32 * dc + 8 * g);
            aQ[mt][dc].u[1] = *(const uint4*)(qrow + 32 * dc + 16 + 8 * g);
        }
    }

    v8f zacc[2][4];
    float mrow[2][8], lrow[2][8];
#pragma unroll
    for (int mt = 0; mt < 2; ++mt) {
#pragma unroll
        for (int j = 0; j < 4; ++j) zacc[mt][j] = v8f_zero();
#pragma unroll
        for (int r = 0; r < 8; ++r) { mrow[mt][r] = -3.0e38f; lrow[mt][r] = 0.f; }
    }

    const u16* vbaseh = vt + (size_t)((b * Hc + h) * HDc) * Tc;

    for (int kv0 = 0; kv0 < q0 + 32; kv0 += 32) {
        // ---- S = Q K^T : two 16-col kv tiles, K operands shared by both M-tiles ----
        v8f s[2][2];
#pragma unroll
        for (int si = 0; si < 2; ++si) {
            const u16* krow =
                qkv + (size_t)(b * Tc + kv0 + 16 * si + nm) * D3 + Dc + h * HDc;
            V16U bk0, bk1;
            bk0.u[0] = *(const uint4*)(krow + 8 * g);
            bk0.u[1] = *(const uint4*)(krow + 16 + 8 * g);
            bk1.u[0] = *(const uint4*)(krow + 32 + 8 * g);
            bk1.u[1] = *(const uint4*)(krow + 32 + 16 + 8 * g);
#pragma unroll
            for (int mt = 0; mt < 2; ++mt) {
                v8f acc = bf16_wmma(aQ[mt][0], bk0, v8f_zero());
                s[mt][si] = bf16_wmma(aQ[mt][1], bk1, acc);
            }
        }

        // ---- causal mask + online softmax, one fused pass per row ----
#pragma unroll
        for (int mt = 0; mt < 2; ++mt) {
#pragma unroll
            for (int r = 0; r < 8; ++r) {
                const int row = q0 + mt * 16 + r + 8 * g;
                const int c0 = kv0 + nm;
                float v0 = s[mt][0][r] * SCALE; if (c0 > row)      v0 = -3.0e38f;
                float v1 = s[mt][1][r] * SCALE; if (c0 + 16 > row) v1 = -3.0e38f;
                float t = fmaxf(v0, v1);
                t = fmaxf(t, __shfl_xor(t, 1, 32));
                t = fmaxf(t, __shfl_xor(t, 2, 32));
                t = fmaxf(t, __shfl_xor(t, 4, 32));
                t = fmaxf(t, __shfl_xor(t, 8, 32));
                float mn = fmaxf(mrow[mt][r], t);
                float p0 = __expf(v0 - mn);
                float p1 = __expf(v1 - mn);
                Pbuf[wave][mt][r + 8 * g][nm]      = f32_to_bf16(p0);
                Pbuf[wave][mt][r + 8 * g][16 + nm] = f32_to_bf16(p1);
                float rs = p0 + p1;
                rs += __shfl_xor(rs, 1, 32);
                rs += __shfl_xor(rs, 2, 32);
                rs += __shfl_xor(rs, 4, 32);
                rs += __shfl_xor(rs, 8, 32);
                float alpha = __expf(mrow[mt][r] - mn);
                lrow[mt][r] = lrow[mt][r] * alpha + rs;
                mrow[mt][r] = mn;
                zacc[mt][0][r] *= alpha; zacc[mt][1][r] *= alpha;
                zacc[mt][2][r] *= alpha; zacc[mt][3][r] *= alpha;
            }
        }
        asm volatile("" ::: "memory");  // order LDS reads after stores (DS is in-order)

        // re-load P tiles in A-matrix layout
        V16U aP[2];
#pragma unroll
        for (int mt = 0; mt < 2; ++mt) {
            aP[mt].u[0] = *(const uint4*)(&Pbuf[wave][mt][nm][8 * g]);
            aP[mt].u[1] = *(const uint4*)(&Pbuf[wave][mt][nm][16 + 8 * g]);
        }

        // ---- Z += P V : V operands (contiguous in t) reused by both M-tiles ----
#pragma unroll
        for (int dt = 0; dt < 4; ++dt) {
            const u16* vrow = vbaseh + (size_t)(dt * 16 + nm) * Tc + kv0;
            V16U bv;
            bv.u[0] = *(const uint4*)(vrow + 8 * g);
            bv.u[1] = *(const uint4*)(vrow + 16 + 8 * g);
            zacc[0][dt] = bf16_wmma(aP[0], bv, zacc[0][dt]);
            zacc[1][dt] = bf16_wmma(aP[1], bv, zacc[1][dt]);
        }
    }

    // ---- finalize: Z /= l ; store bf16 into z[b, t, h*64 + d] ----
#pragma unroll
    for (int mt = 0; mt < 2; ++mt)
#pragma unroll
        for (int dt = 0; dt < 4; ++dt)
#pragma unroll
            for (int r = 0; r < 8; ++r) {
                size_t idx = (size_t)(b * Tc + q0 + mt * 16 + r + 8 * g) * Dc +
                             h * HDc + dt * 16 + nm;
                zout[idx] = f32_to_bf16(zacc[mt][dt][r] / lrow[mt][r]);
            }
}

// ---------------- workspace layout (bytes) ----------------
// Xbf   @ 0          :  8,388,608
// Wqkv  @ 8388608    :  6,291,456
// Wout  @ 14680064   :  2,097,152
// QKV   @ 16777216   : 25,165,824
// Vt    @ 41943040   :  8,388,608
// Zbf   @ 50331648   :  8,388,608   -> total 56 MiB

extern "C" void kernel_launch(void* const* d_in, const int* in_sizes, int n_in,
                              void* d_out, int out_size, void* d_ws, size_t ws_size,
                              hipStream_t stream) {
    const float* X    = (const float*)d_in[0];
    const float* Wqkv = (const float*)d_in[1];
    const float* Wout = (const float*)d_in[2];
    float* out = (float*)d_out;

    char* ws = (char*)d_ws;
    u16* Xbf    = (u16*)(ws + 0);
    u16* Wqkvbf = (u16*)(ws + 8388608);
    u16* Woutbf = (u16*)(ws + 14680064);
    u16* QKVbf  = (u16*)(ws + 16777216);
    u16* Vtbf   = (u16*)(ws + 41943040);
    u16* Zbf    = (u16*)(ws + 50331648);

    const unsigned nX  = (unsigned)Bc * Tc * Dc;  // 4,194,304
    const unsigned nWq = (unsigned)D3 * Dc;       // 3,145,728
    const unsigned nWo = (unsigned)Dc * Dc;       // 1,048,576

    cvt_f32_bf16_kernel<<<4096, 256, 0, stream>>>(X, Xbf, nX);
    cvt_f32_bf16_kernel<<<4096, 256, 0, stream>>>(Wqkv, Wqkvbf, nWq);
    cvt_f32_bf16_kernel<<<2048, 256, 0, stream>>>(Wout, Woutbf, nWo);

    // QKV = X @ Wqkv^T : M=4096, N=3072, K=1024 -> (128*48)=6144 waves / 8 = 768 blocks
    gemm_wmma_bf16<true><<<768, 256, 0, stream>>>(Xbf, Wqkvbf, (void*)QKVbf,
                                                  Bc * Tc, D3, Dc);

    vtrans_kernel<<<4096, 256, 0, stream>>>(QKVbf, Vtbf);

    // flash attention: 2*16*64 = 2048 waves / 8 = 256 blocks
    flash_attn_kernel<<<256, 256, 0, stream>>>(QKVbf, Vtbf, Zbf);

    // out = Z @ Wout^T : M=4096, N=1024, K=1024 -> (128*16)=2048 waves / 8 = 256 blocks
    gemm_wmma_bf16<false><<<256, 256, 0, stream>>>(Zbf, Woutbf, (void*)out,
                                                   Bc * Tc, Dc, Dc);
}
